// MambaBlock_59176059404396
// MI455X (gfx1250) — matrix operand order repaired and checked by reading, not compiled, not run on hardware
//
#include <hip/hip_runtime.h>
#include <hip/hip_bf16.h>

// ---------------------------------------------------------------------------
// Mamba block forward for MI455X (gfx1250, wave32).
// GEMMs: v_wmma_f32_16x16x32_bf16, A-tiles staged to LDS by the Tensor Data
// Mover (tensor_load_to_lds + s_wait_tensorcnt), double buffered.
// ---------------------------------------------------------------------------

typedef __attribute__((ext_vector_type(16))) __bf16 v16bf;
typedef __attribute__((ext_vector_type(8)))  __bf16 bf16x8;
typedef __attribute__((ext_vector_type(4)))  __bf16 bf16x4;
typedef __attribute__((ext_vector_type(8)))  float  v8f;
typedef __attribute__((ext_vector_type(4)))  unsigned int u32x4;
typedef __attribute__((ext_vector_type(8)))  int i32x8;
typedef __attribute__((ext_vector_type(4)))  int i32x4;

#define D_MODEL 1024
#define D_INNER 2048
#define DT_RANK 64
#define D_STATE 16
#define BATCH   4
#define SEQ     2048
#define ROWS    (BATCH * SEQ)          // 8192

#if defined(__has_builtin)
#if __has_builtin(__builtin_amdgcn_tensor_load_to_lds)
#define HAVE_TDM 1
#endif
#endif
#ifndef HAVE_TDM
#define HAVE_TDM 0
#endif

__device__ __forceinline__ float fast_silu(float x) {
  return x * (1.0f / (1.0f + __expf(-x)));
}

#if HAVE_TDM
// 2-D tile DMA: global (bf16, row pitch = stride0 elems) -> LDS (packed rows).
// D# per CDNA5 ISA 08_async_tensor §8.3-8.5: group0 = {count=1, lds_addr,
// global_addr, type=2}; group1 = {data_size=1(2B), tensor dims, tile dims,
// dim0 stride}; groups 2/3 zero (2-D tensor).
__device__ __forceinline__ void tdm_load_2d(unsigned int lds_off,
                                            unsigned long long gaddr,
                                            unsigned int tdim0, unsigned int tdim1,
                                            unsigned int tile0, unsigned int tile1,
                                            unsigned long long stride0) {
  u32x4 g0;
  g0[0] = 1u;                                            // count=1, user mode
  g0[1] = lds_off;                                       // LDS byte address
  g0[2] = (unsigned int)(gaddr & 0xffffffffull);         // global_addr[31:0]
  g0[3] = (unsigned int)((gaddr >> 32) & 0x1ffffffull)   // global_addr[56:32]
        | (2u << 30);                                    // type=2 ("image")
  i32x8 g1;
  g1[0] = (int)(1u << 16);                               // data_size = 1 -> 2B
  g1[1] = (int)((tdim0 & 0xffffu) << 16);                // tensor_dim0[15:0]
  g1[2] = (int)((tdim0 >> 16) | ((tdim1 & 0xffffu) << 16));
  g1[3] = (int)((tdim1 >> 16) | (tile0 << 16));          // tile_dim0
  g1[4] = (int)(tile1 & 0xffffu);                        // tile_dim1 (tile_dim2=0)
  g1[5] = (int)(unsigned int)(stride0 & 0xffffffffull);  // dim0 stride lo
  g1[6] = (int)(unsigned int)((stride0 >> 32) & 0xffffull);
  g1[7] = 0;
  i32x4 z4 = {0, 0, 0, 0};
#if defined(__clang_major__) && (__clang_major__ >= 23)
  i32x8 z8 = {0, 0, 0, 0, 0, 0, 0, 0};
  __builtin_amdgcn_tensor_load_to_lds(g0, g1, z4, z4, z8, 0);
#else
  __builtin_amdgcn_tensor_load_to_lds(g0, g1, z4, z4, 0);
#endif
}
#endif

// --- WMMA fragment loaders (layouts per CDNA5 ISA 7.12.2, wave32) -----------

// A fragment 16(M)x32(K) from LDS tile with packed 32-elem rows.
// lanes 0-15: M=lane, elems 0-7 = K 0..7, 8-15 = K 16..23; lanes 16-31: +8 K.
__device__ __forceinline__ v16bf load_a_frag_lds(const __bf16* __restrict__ tile,
                                                 int mrow) {
  const int lane = threadIdx.x & 31;
  const int m  = mrow + (lane & 15);
  const int kb = (lane >> 4) << 3;          // 0 or 8
  const __bf16* p = tile + m * 32 + kb;
  bf16x8 lo = *(const bf16x8*)p;            // K kb..kb+7
  bf16x8 hi = *(const bf16x8*)(p + 16);     // K kb+16..kb+23
  return __builtin_shufflevector(lo, hi, 0, 1, 2, 3, 4, 5, 6, 7,
                                         8, 9, 10, 11, 12, 13, 14, 15);
}

// A fragment direct from global bf16 (fallback when TDM builtin is absent).
__device__ __forceinline__ v16bf load_a_frag_g(const __bf16* __restrict__ A, int lda) {
  const int lane = threadIdx.x & 31;
  const int m  = lane & 15;
  const int kb = (lane >> 4) << 3;
  const __bf16* p = A + (size_t)m * lda + kb;
  bf16x8 lo = *(const bf16x8*)p;
  bf16x8 hi = *(const bf16x8*)(p + 16);
  return __builtin_shufflevector(lo, hi, 0, 1, 2, 3, 4, 5, 6, 7,
                                         8, 9, 10, 11, 12, 13, 14, 15);
}

// B fragment 32(K)x16(N) from ROW-MAJOR bf16 weight W (N x K): B[k][n]=W[n][k].
// lanes 0-15: elems 0-7 = K 0..7, 8-15 = K 16..23; lanes 16-31: +8 K.
__device__ __forceinline__ v16bf load_b_frag(const __bf16* __restrict__ W, int ldw) {
  const int lane = threadIdx.x & 31;
  const int n  = lane & 15;
  const int kb = (lane >> 4) << 3;          // 0 or 8
  const __bf16* p = W + (size_t)n * ldw + kb;
  bf16x8 lo = *(const bf16x8*)p;
  bf16x8 hi = *(const bf16x8*)(p + 16);
  return __builtin_shufflevector(lo, hi, 0, 1, 2, 3, 4, 5, 6, 7,
                                         8, 9, 10, 11, 12, 13, 14, 15);
}

// ---------------------------------------------------------------------------
// WMMA GEMM:  C[M,N] = epi( A[M,K]_bf16 x W[N,K]_bf16^T )  (f32 accumulate)
// Wave tile (16*TM)x(16*TN); WM x WN waves per block; exact tiling everywhere
// so EXEC is all ones at each WMMA.  A block tile (BM x 32) is DMA'd to LDS
// by the TDM (wave 0), double buffered; B fragments stream from global.
// EPI: 0 = none, 1 = +bias[col] then softplus (dt path).
// ---------------------------------------------------------------------------
template <int TM, int TN, int WM, int WN, int EPI>
__global__ __launch_bounds__(WM * WN * 32)
void gemm_wmma(const __bf16* __restrict__ A, const __bf16* __restrict__ W,
               const float* __restrict__ bias, float* __restrict__ C,
               int K, int lda, int ldw, int ldc) {
  constexpr int BM = 16 * TM * WM;
  const int wave = threadIdx.x >> 5;
  const int wm = wave / WN;
  const int wn = wave % WN;
  const int rowW = wm * (16 * TM);                      // wave row in block
  const int row0 = blockIdx.y * BM + rowW;
  const int col0 = blockIdx.x * (16 * TN * WN) + wn * (16 * TN);

  v8f acc[TM][TN] = {};

#if HAVE_TDM
  __shared__ alignas(16) __bf16 atile[2][BM * 32];
  const unsigned long long abase =
      (unsigned long long)(uintptr_t)A +
      (unsigned long long)((size_t)blockIdx.y * BM) * (unsigned long long)lda * 2ull;
  if (threadIdx.x < 32)   // wave 0 drives the TDM (EXEC ignored by TDM)
    tdm_load_2d(0u, abase, (unsigned)lda, (unsigned)BM, 32u, (unsigned)BM,
                (unsigned long long)lda);
  int pb = 0;
#endif

  for (int k = 0; k < K; k += 32) {
    const bool hasNext = (k + 32 < K);
#if HAVE_TDM
    if (threadIdx.x < 32) {
      if (hasNext)
        tdm_load_2d((unsigned)((pb ^ 1) * BM * 32 * 2),
                    abase + (unsigned long long)(k + 32) * 2ull,
                    (unsigned)lda, (unsigned)BM, 32u, (unsigned)BM,
                    (unsigned long long)lda);
      // TDM ops complete in order per wave: <=1 leaves next outstanding.
      if (hasNext) __builtin_amdgcn_s_wait_tensorcnt(1);
      else         __builtin_amdgcn_s_wait_tensorcnt(0);
    }
    __syncthreads();      // current LDS tile visible to all waves
#endif
    if (hasNext)
      __builtin_prefetch(W + (size_t)col0 * ldw + (k + 32), 0, 1);

    v16bf af[TM];
#if HAVE_TDM
#pragma unroll
    for (int i = 0; i < TM; ++i)
      af[i] = load_a_frag_lds(&atile[pb][0], rowW + 16 * i);
#else
#pragma unroll
    for (int i = 0; i < TM; ++i)
      af[i] = load_a_frag_g(A + (size_t)(row0 + 16 * i) * lda + k, lda);
#endif
    v16bf bfr[TN];
#pragma unroll
    for (int j = 0; j < TN; ++j)
      bfr[j] = load_b_frag(W + (size_t)(col0 + 16 * j) * ldw + k, ldw);

#pragma unroll
    for (int i = 0; i < TM; ++i)
#pragma unroll
      for (int j = 0; j < TN; ++j)
        acc[i][j] = __builtin_amdgcn_wmma_f32_16x16x32_bf16(
            false, af[i], false, bfr[j], (short)0, acc[i][j], false, false);

#if HAVE_TDM
    __syncthreads();      // all reads of atile[pb] done before its reuse
    pb ^= 1;
#endif
  }

  // C/D layout: VGPR r, lanes 0-15 -> (M=r, N=lane); lanes 16-31 -> (M=8+r).
  const int lane = threadIdx.x & 31;
  const int n  = lane & 15;
  const int mb = (lane >> 4) << 3;
#pragma unroll
  for (int i = 0; i < TM; ++i)
#pragma unroll
    for (int j = 0; j < TN; ++j) {
      const int col = col0 + 16 * j + n;
      float* cp = C + (size_t)(row0 + 16 * i + mb) * ldc + col;
#pragma unroll
      for (int r = 0; r < 8; ++r) {
        float v = acc[i][j][r];
        if (EPI == 1) {                       // bias + softplus (dt path)
          v += bias[col];
          v = (v > 20.0f) ? v : __logf(1.0f + __expf(v));
        }
        cp[(size_t)r * ldc] = v;
      }
    }
}

// ---------------------------------------------------------------------------
// f32 -> bf16 pre-conversion (n must be a multiple of 1024; 4 elems/thread).
// ---------------------------------------------------------------------------
__global__ __launch_bounds__(256)
void cvt_bf16_kernel(const float* __restrict__ in, __bf16* __restrict__ out) {
  const int i = (blockIdx.x * 256 + threadIdx.x) * 4;
  const float4 v = *(const float4*)(in + i);
  bf16x4 o;
  o[0] = (__bf16)v.x; o[1] = (__bf16)v.y; o[2] = (__bf16)v.z; o[3] = (__bf16)v.w;
  *(bf16x4*)(out + i) = o;
}

// Extract dt columns (0..63) of x_dbl (ld 96) into a packed bf16 (ROWS x 64).
__global__ __launch_bounds__(256)
void extract_dt_kernel(const float* __restrict__ xdbl, __bf16* __restrict__ out) {
  const int idx = blockIdx.x * 256 + threadIdx.x;       // over ROWS*64
  const int r = idx >> 6, c = idx & 63;
  out[idx] = (__bf16)xdbl[(size_t)r * 96 + c];
}

// ---------------------------------------------------------------------------
// Causal depthwise conv1d (width 4, left pad 3) + bias + SiLU.
// Writes u in f32 (for the scan) and bf16 (GEMM operand).
// ---------------------------------------------------------------------------
__global__ __launch_bounds__(256)
void conv_silu_kernel(const float* __restrict__ xr, const float* __restrict__ cw,
                      const float* __restrict__ cb, float* __restrict__ u,
                      __bf16* __restrict__ ubf) {
  const int idx = blockIdx.x * 256 + threadIdx.x;       // over ROWS*D_INNER
  const int c   = idx & (D_INNER - 1);
  const int row = idx >> 11;                            // b*SEQ + l
  const int l   = row & (SEQ - 1);
  float acc = cb[c];
#pragma unroll
  for (int j = 0; j < 4; ++j) {
    const int ls = l - 3 + j;
    if (ls >= 0)
      acc += xr[(size_t)(row - l + ls) * (2 * D_INNER) + c] * cw[c * 4 + j];
  }
  const float s = fast_silu(acc);
  u[idx]   = s;
  ubf[idx] = (__bf16)s;
}

// ---------------------------------------------------------------------------
// Selective scan: one thread per (batch, channel); 16 states in registers.
// Fuses the output gate: emits ygated = y * silu(res) directly in bf16.
// ---------------------------------------------------------------------------
__global__ __launch_bounds__(256)
void scan_kernel(const float* __restrict__ u, const float* __restrict__ dts,
                 const float* __restrict__ xdbl, const float* __restrict__ A_log,
                 const float* __restrict__ Dv, const float* __restrict__ xr,
                 __bf16* __restrict__ ybf) {
  const int c = blockIdx.x * 256 + threadIdx.x;   // channel
  const int b = blockIdx.y;                       // batch

  float Ac[D_STATE];
#pragma unroll
  for (int s = 0; s < D_STATE; ++s) Ac[s] = -__expf(A_log[c * D_STATE + s]);
  const float Dc = Dv[c];

  float h[D_STATE] = {};

  const float* up = u    + (size_t)b * SEQ * D_INNER + c;
  const float* dp = dts  + (size_t)b * SEQ * D_INNER + c;
  const float* xb = xdbl + (size_t)b * SEQ * 96;
  const float* rp = xr   + (size_t)b * SEQ * (2 * D_INNER) + D_INNER + c;  // res
  __bf16*      yp = ybf  + (size_t)b * SEQ * D_INNER + c;

  for (int l = 0; l < SEQ; ++l) {
    const float ut = up[(size_t)l * D_INNER];
    const float dt = dp[(size_t)l * D_INNER];

    const float4* Bp = (const float4*)(xb + (size_t)l * 96 + 64);  // block-uniform
    const float4* Cp = (const float4*)(xb + (size_t)l * 96 + 80);
    float Bl[D_STATE], Cl[D_STATE];
#pragma unroll
    for (int q = 0; q < 4; ++q) {
      float4 bq = Bp[q], cq = Cp[q];
      Bl[4*q+0] = bq.x; Bl[4*q+1] = bq.y; Bl[4*q+2] = bq.z; Bl[4*q+3] = bq.w;
      Cl[4*q+0] = cq.x; Cl[4*q+1] = cq.y; Cl[4*q+2] = cq.z; Cl[4*q+3] = cq.w;
    }

    const float du = dt * ut;
    float acc = Dc * ut;
#pragma unroll
    for (int s = 0; s < D_STATE; ++s) {
      const float dA = __expf(dt * Ac[s]);
      h[s] = dA * h[s] + du * Bl[s];
      acc += h[s] * Cl[s];
    }
    const float res = rp[(size_t)l * (2 * D_INNER)];
    yp[(size_t)l * D_INNER] = (__bf16)(acc * fast_silu(res));
  }
}

// ---------------------------------------------------------------------------
extern "C" void kernel_launch(void* const* d_in, const int* in_sizes, int n_in,
                              void* d_out, int out_size, void* d_ws, size_t ws_size,
                              hipStream_t stream) {
  const float* x        = (const float*)d_in[0];   // (4,2048,1024)
  const float* in_proj  = (const float*)d_in[1];   // (4096,1024)
  const float* conv_w   = (const float*)d_in[2];   // (2048,1,4)
  const float* conv_b   = (const float*)d_in[3];   // (2048,)
  const float* x_proj   = (const float*)d_in[4];   // (96,2048)
  const float* dt_proj  = (const float*)d_in[5];   // (2048,64)
  const float* dt_b     = (const float*)d_in[6];   // (2048,)
  const float* A_log    = (const float*)d_in[7];   // (2048,16)
  const float* Dv       = (const float*)d_in[8];   // (2048,)
  const float* out_proj = (const float*)d_in[9];   // (1024,2048)
  float* out = (float*)d_out;                      // (4,2048,1024)

  char* wsp = (char*)d_ws;
  auto alloc = [&](size_t bytes) -> char* {
    char* p = wsp;
    wsp += (bytes + 255) & ~(size_t)255;
    return p;
  };
  float*  xr   = (float*)alloc((size_t)ROWS * 4096 * 4);      // xs | res
  float*  u    = (float*)alloc((size_t)ROWS * D_INNER * 4);
  float*  dts  = (float*)alloc((size_t)ROWS * D_INNER * 4);
  float*  xdbl = (float*)alloc((size_t)ROWS * 96 * 4);        // dt_raw | B | C
  __bf16* xbf  = (__bf16*)alloc((size_t)ROWS * D_MODEL * 2);
  __bf16* ubf  = (__bf16*)alloc((size_t)ROWS * D_INNER * 2);
  __bf16* ybf  = (__bf16*)alloc((size_t)ROWS * D_INNER * 2);
  __bf16* dtbf = (__bf16*)alloc((size_t)ROWS * DT_RANK * 2);
  __bf16* wbf1 = (__bf16*)alloc((size_t)(2 * D_INNER) * D_MODEL * 2);
  __bf16* wbf3 = (__bf16*)alloc((size_t)96 * D_INNER * 2);
  __bf16* wbf4 = (__bf16*)alloc((size_t)D_INNER * DT_RANK * 2);
  __bf16* wbf6 = (__bf16*)alloc((size_t)D_MODEL * D_INNER * 2);

  // 0) pre-convert activations/weights to bf16 (memory-bound, ~60MB total)
  cvt_bf16_kernel<<<(ROWS * D_MODEL) / 1024, 256, 0, stream>>>(x, xbf);
  cvt_bf16_kernel<<<(2 * D_INNER * D_MODEL) / 1024, 256, 0, stream>>>(in_proj, wbf1);
  cvt_bf16_kernel<<<(96 * D_INNER) / 1024, 256, 0, stream>>>(x_proj, wbf3);
  cvt_bf16_kernel<<<(D_INNER * DT_RANK) / 1024, 256, 0, stream>>>(dt_proj, wbf4);
  cvt_bf16_kernel<<<(D_MODEL * D_INNER) / 1024, 256, 0, stream>>>(out_proj, wbf6);

  // 1) x_and_res = x @ in_proj^T          (8192x1024)x(1024x4096)
  gemm_wmma<4, 4, 2, 4, 0><<<dim3(4096 / 256, ROWS / 128), 256, 0, stream>>>(
      xbf, wbf1, nullptr, xr, D_MODEL, D_MODEL, D_MODEL, 2 * D_INNER);

  // 2) u = silu(depthwise_causal_conv(xs) + b)   (f32 + bf16 copies)
  conv_silu_kernel<<<(ROWS * D_INNER) / 256, 256, 0, stream>>>(xr, conv_w, conv_b, u, ubf);

  // 3) x_dbl = u @ x_proj^T               (8192x2048)x(2048x96)
  gemm_wmma<4, 3, 4, 2, 0><<<dim3(96 / 96, ROWS / 256), 256, 0, stream>>>(
      ubf, wbf3, nullptr, xdbl, D_INNER, D_INNER, D_INNER, 96);

  // 3b) pack dt columns to bf16
  extract_dt_kernel<<<(ROWS * DT_RANK) / 256, 256, 0, stream>>>(xdbl, dtbf);

  // 4) dt_s = softplus(dt_raw @ dt_proj^T + dt_b)
  gemm_wmma<4, 4, 2, 4, 1><<<dim3(D_INNER / 256, ROWS / 128), 256, 0, stream>>>(
      dtbf, wbf4, dt_b, dts, DT_RANK, DT_RANK, DT_RANK, D_INNER);

  // 5) selective scan + fused output gate -> ygated (bf16)
  scan_kernel<<<dim3(D_INNER / 256, BATCH), 256, 0, stream>>>(
      u, dts, xdbl, A_log, Dv, xr, ybf);

  // 6) out = ygated @ out_proj^T          (8192x2048)x(2048x1024)
  gemm_wmma<4, 4, 2, 4, 0><<<dim3(D_MODEL / 256, ROWS / 128), 256, 0, stream>>>(
      ybf, wbf6, nullptr, out, D_INNER, D_INNER, D_INNER, D_MODEL);
}
